// DinMod_23957327577772
// MI455X (gfx1250) — matrix operand order repaired and checked by reading, not compiled, not run on hardware
//
#include <hip/hip_runtime.h>
#include <hip/hip_bf16.h>

// CfC recurrent network, MI455X (gfx1250), wave32 + WMMA f16 16x16x32.
// B=256, T=512, IN=64, LATENT=256, BACKBONE=128, OUT=64.

typedef __attribute__((ext_vector_type(16))) _Float16 v16h;
typedef __attribute__((ext_vector_type(8)))  _Float16 v8h;
typedef __attribute__((ext_vector_type(8)))  float    v8f;
typedef __attribute__((ext_vector_type(4)))  float    v4f;

#define CFC_B        256
#define CFC_T        512
#define CFC_IN       64
#define CFC_LAT      256
#define CFC_BB       128
#define CFC_OUT      64

// packed fragment sizes (halves). fragment = 512 halves (32 lanes x 16).
#define WB_FRAGS     (10 * 8)        // K=320 (10 ktiles) x N=128 (8 ntiles)
#define HEAD_FRAGS   (4 * 48)        // K=128 x N=768 (W1|W2|Wa+Wtb)
#define WP1_FRAGS    (8 * 8)         // K=256 x N=128
#define WP2_FRAGS    (4 * 4)         // K=128 x N=64
#define WB_HALVES    (WB_FRAGS * 512)
#define HEAD_HALVES  (HEAD_FRAGS * 512)
#define WP1_HALVES   (WP1_FRAGS * 512)
#define WP2_HALVES   (WP2_FRAGS * 512)

// LDS layout (halves): heads weights + h/z/hdn scratch. Wb lives in VGPRs.
#define LDS_HEAD_OFF  0
#define LDS_H_OFF     (LDS_HEAD_OFF + HEAD_HALVES)      // 98304
#define LDS_Z_OFF     (LDS_H_OFF + 16 * CFC_LAT)        // +4096
#define LDS_HDN_OFF   (LDS_Z_OFF + 16 * CFC_BB)         // +2048
#define LDS_TOT_HALVES (LDS_HDN_OFF + 16 * CFC_BB)      // 106496
#define LDS_BYTES     (LDS_TOT_HALVES * 2)              // 212992 B <= 320KB

#define LOG2E 1.4426950408889634f

static __device__ __forceinline__ float fast_tanh(float x) {
#if __has_builtin(__builtin_amdgcn_tanhf)
  return __builtin_amdgcn_tanhf(x);                    // v_tanh_f32 (CDNA5 TRANS)
#elif __has_builtin(__builtin_amdgcn_tanh_f32)
  return __builtin_amdgcn_tanh_f32(x);
#else
  float cx = fminf(fmaxf(x, -9.0f), 9.0f);
  float e = __builtin_amdgcn_exp2f(cx * (2.0f * LOG2E));
  return (e - 1.0f) * __builtin_amdgcn_rcpf(e + 1.0f);
#endif
}

static __device__ __forceinline__ float fast_sigmoid(float x) {
  // 1/(1+exp(-x)) via native v_exp_f32 (2^x) + v_rcp_f32
  return __builtin_amdgcn_rcpf(1.0f + __builtin_amdgcn_exp2f(-x * LOG2E));
}

static __device__ __forceinline__ v8f wmma16x16x32(v16h a, v16h b, v8f c) {
  return __builtin_amdgcn_wmma_f32_16x16x32_f16(
      /*neg_a=*/false, a, /*neg_b=*/false, b,
      /*c_mod=*/(short)0, c, /*reuse_a=*/false, /*reuse_b=*/false);
}

// Load a pre-packed fragment: 16 contiguous halves for this lane.
static __device__ __forceinline__ v16h ldfrag(const _Float16* p, int lane) {
  const _Float16* q = p + lane * 16;
  v8h lo = *(const v8h*)q;
  v8h hi = *(const v8h*)(q + 8);
  v16h r;
#pragma unroll
  for (int i = 0; i < 8; ++i) { r[i] = lo[i]; r[8 + i] = hi[i]; }
  return r;
}

// Load an A fragment (16x32 f16) from a row-major f16 buffer.
// rowptr must point at (row = lane&15)*ld + kt*32 + 8*(lane>>4).
static __device__ __forceinline__ v16h ldA(const _Float16* rowptr) {
  v8h lo = *(const v8h*)rowptr;          // K = koff .. koff+7
  v8h hi = *(const v8h*)(rowptr + 16);   // K = koff+16 .. koff+23
  v16h r;
#pragma unroll
  for (int i = 0; i < 8; ++i) { r[i] = lo[i]; r[8 + i] = hi[i]; }
  return r;
}

// ---------------- prep: pack weights into WMMA B-fragment layout -------------
// B fragment layout (16x16x32 f16): lane n=(l&15) holds column nt*16+n;
// element e holds K = kt*32 + 8*(l>>4) + (e&7) + 16*(e>>3).
__global__ void pack_b_f16(const float* __restrict__ W, _Float16* __restrict__ dst,
                           int K, int N) {
  int idx = blockIdx.x * blockDim.x + threadIdx.x;
  int total = (K >> 5) * (N >> 4) * 512;
  if (idx >= total) return;
  int e = idx & 15;
  int lane = (idx >> 4) & 31;
  int frag = idx >> 9;
  int ntiles = N >> 4;
  int nt = frag % ntiles;
  int kt = frag / ntiles;
  int n = nt * 16 + (lane & 15);
  int k = kt * 32 + 8 * (lane >> 4) + (e & 7) + ((e >> 3) << 4);
  dst[idx] = (_Float16)W[k * N + n];
}

// fused heads: virtual weight [128 x 768] = [W1 | W2 | (Wa+Wtb)]
__global__ void pack_heads_f16(const float* __restrict__ W1, const float* __restrict__ W2,
                               const float* __restrict__ Wa, const float* __restrict__ Wtb,
                               _Float16* __restrict__ dst) {
  int idx = blockIdx.x * blockDim.x + threadIdx.x;
  if (idx >= HEAD_HALVES) return;
  int e = idx & 15;
  int lane = (idx >> 4) & 31;
  int frag = idx >> 9;
  int nt = frag % 48;
  int kt = frag / 48;
  int n = nt * 16 + (lane & 15);
  int k = kt * 32 + 8 * (lane >> 4) + (e & 7) + ((e >> 3) << 4);
  float v;
  if (n < 256)      v = W1[k * 256 + n];
  else if (n < 512) v = W2[k * 256 + (n - 256)];
  else              v = Wa[k * 256 + (n - 512)] + Wtb[k * 256 + (n - 512)];
  dst[idx] = (_Float16)v;
}

// ---------------- fused recurrent scan + projection --------------------------
__global__ __launch_bounds__(256) void cfc_scan_kernel(
    const float* __restrict__ x,          // [256,512,64] f32
    const float* __restrict__ bbv,        // [128]
    const float* __restrict__ b1v,        // [256]
    const float* __restrict__ b2v,        // [256]
    const float* __restrict__ bav,        // [256]
    const float* __restrict__ btbv,       // [256]
    const float* __restrict__ bp1v,       // [128]
    const float* __restrict__ bp2v,       // [64]
    const _Float16* __restrict__ wbPack,  // 80 frags  -> hoisted to VGPRs
    const _Float16* __restrict__ headPack,// 192 frags -> LDS resident
    const _Float16* __restrict__ wp1Pack, // 64 frags  (global/L2 hot)
    const _Float16* __restrict__ wp2Pack, // 16 frags  (global/L2 hot)
    float* __restrict__ out)              // [256,512,64] f32
{
  extern __shared__ _Float16 lds[];
  _Float16* sHead = lds + LDS_HEAD_OFF;
  _Float16* sH    = lds + LDS_H_OFF;     // 16 x 256 row-major f16
  _Float16* sZ    = lds + LDS_Z_OFF;     // 16 x 128
  _Float16* sHdn  = lds + LDS_HDN_OFF;   // 16 x 128

  const int tid   = threadIdx.x;
  const int lane  = tid & 31;
  const int wave  = tid >> 5;            // 0..7 (wave32)
  const int mtile = blockIdx.x;          // 0..15 : 16 batch rows each
  const int row   = lane & 15;
  const int hiK   = (lane >> 4) * 8;     // A-fragment lane-half K offset
  const int m0    = (lane >> 4) * 8;     // D-fragment row base for this lane

  // stage fused-head weights into LDS once; zero h
  {
    const v8h* s1 = (const v8h*)headPack;
    v8h* d1 = (v8h*)sHead;
    for (int i = tid; i < HEAD_HALVES / 8; i += 256) d1[i] = s1[i];
    v8h zz = {};
    v8h* dh = (v8h*)sH;
    for (int i = tid; i < (16 * CFC_LAT) / 8; i += 256) dh[i] = zz;
  }

  // hoist this wave's backbone B-fragments (constant over t) into registers
  v16h wbf[10];
#pragma unroll
  for (int kt = 0; kt < 10; ++kt)
    wbf[kt] = ldfrag(wbPack + (kt * 8 + wave) * 512, lane);

  // persistent A-fragments of h (shared by wp1 at step t and backbone at t+1).
  // h0 == 0, so initialize without loads.
  v16h ah[8];
#pragma unroll
  for (int kt = 0; kt < 8; ++kt) {
    v16h zfrag = {};
    ah[kt] = zfrag;
  }

  // loop-invariant biases (per lane)
  const int ncolB = wave * 16 + (lane & 15);          // backbone / wp1 column
  const float biasZ  = bbv[ncolB];
  const float biasP1 = bp1v[ncolB];
  float c1h[2], c2h[2], c3h[2];
#pragma unroll
  for (int half = 0; half < 2; ++half) {
    int ncol = (wave + half * 8) * 16 + (lane & 15);  // heads latent column
    c1h[half] = b1v[ncol];
    c2h[half] = b2v[ncol];
    c3h[half] = bav[ncol] + btbv[ncol];
  }
  float biasP2 = 0.0f;
  if (wave < 4) biasP2 = bp2v[wave * 16 + (lane & 15)];

  __syncthreads();

  const float* xrow = x + (size_t)(mtile * 16 + row) * CFC_T * CFC_IN;

  for (int t = 0; t < CFC_T; ++t) {
    const float* xr = xrow + t * CFC_IN;
    if (t + 1 < CFC_T) __builtin_prefetch(xrow + (t + 1) * CFC_IN, 0, 1);

    // ---- backbone: z = lecun_tanh([x_t|h] @ Wb + bb); wave owns ntile=wave
    // A for the h-part is register-resident (ah[], loaded by wp1 last step),
    // B is register-resident (wbf[]) -> pure WMMA chain, no LDS on this stage.
    v8f acc = {};
#pragma unroll
    for (int kt = 0; kt < 2; ++kt) {               // x part, K = 0..63
      int k0 = kt * 32 + hiK;
      v4f f0 = *(const v4f*)(xr + k0);
      v4f f1 = *(const v4f*)(xr + k0 + 4);
      v4f f2 = *(const v4f*)(xr + k0 + 16);
      v4f f3 = *(const v4f*)(xr + k0 + 20);
      v16h a;
#pragma unroll
      for (int i = 0; i < 4; ++i) {
        a[i] = (_Float16)f0[i]; a[4 + i] = (_Float16)f1[i];
        a[8 + i] = (_Float16)f2[i]; a[12 + i] = (_Float16)f3[i];
      }
      acc = wmma16x16x32(a, wbf[kt], acc);
    }
#pragma unroll
    for (int kt = 0; kt < 8; ++kt)                 // h part, K = 64..319
      acc = wmma16x16x32(ah[kt], wbf[kt + 2], acc);
    {
#pragma unroll
      for (int r = 0; r < 8; ++r) {
        float zv = 1.7159f * fast_tanh(0.666f * (acc[r] + biasZ));
        sZ[(m0 + r) * CFC_BB + ncolB] = (_Float16)zv;
      }
    }
    __syncthreads();   // B1: z visible

    // ---- heads: ff1/ff2/t_interp; wave owns latent n-tiles {wave, wave+8}
    v16h az[4];
#pragma unroll
    for (int kt = 0; kt < 4; ++kt)
      az[kt] = ldA(sZ + row * CFC_BB + kt * 32 + hiK);
#pragma unroll
    for (int half = 0; half < 2; ++half) {
      int lt = wave + half * 8;                    // latent n-tile 0..15
      v8f a1 = {}, a2 = {}, a3 = {};
#pragma unroll
      for (int kt = 0; kt < 4; ++kt) {
        v16h bf1 = ldfrag(sHead + (kt * 48 + lt) * 512, lane);
        v16h bf2 = ldfrag(sHead + (kt * 48 + 16 + lt) * 512, lane);
        v16h bf3 = ldfrag(sHead + (kt * 48 + 32 + lt) * 512, lane);
        a1 = wmma16x16x32(az[kt], bf1, a1);
        a2 = wmma16x16x32(az[kt], bf2, a2);
        a3 = wmma16x16x32(az[kt], bf3, a3);
      }
      int ncol = lt * 16 + (lane & 15);
#pragma unroll
      for (int r = 0; r < 8; ++r) {
        float f1 = fast_tanh(a1[r] + c1h[half]);
        float f2 = fast_tanh(a2[r] + c2h[half]);
        float ti = fast_sigmoid(a3[r] + c3h[half]);
        float hn = f1 + ti * (f2 - f1);
        sH[(m0 + r) * CFC_LAT + ncol] = (_Float16)hn;
      }
    }
    __syncthreads();   // B2: h_new visible

    // ---- projection stage 1: hdn = silu(h @ Wp1 + bp1); wave owns ntile=wave
    // Reload ah[] (grouped, pipelined) -- reused by backbone at t+1.
    {
#pragma unroll
      for (int kt = 0; kt < 8; ++kt)
        ah[kt] = ldA(sH + row * CFC_LAT + kt * 32 + hiK);
      v8f pacc = {};
#pragma unroll
      for (int kt = 0; kt < 8; ++kt) {
        v16h b = ldfrag(wp1Pack + (kt * 8 + wave) * 512, lane);
        pacc = wmma16x16x32(ah[kt], b, pacc);
      }
#pragma unroll
      for (int r = 0; r < 8; ++r) {
        float v = pacc[r] + biasP1;
        float sv = v * fast_sigmoid(v);
        sHdn[(m0 + r) * CFC_BB + ncolB] = (_Float16)sv;
      }
    }
    __syncthreads();   // B3: hdn visible
                       // (no 4th barrier: all next-iteration LDS writes are
                       //  ordered behind B1/B2 of iteration t+1.)

    // ---- projection stage 2: out = hdn @ Wp2 + bp2; waves 0..3 (uniform EXEC)
    if (wave < 4) {
      v8f oacc = {};
#pragma unroll
      for (int kt = 0; kt < 4; ++kt) {
        v16h a = ldA(sHdn + row * CFC_BB + kt * 32 + hiK);
        v16h b = ldfrag(wp2Pack + (kt * 4 + wave) * 512, lane);
        oacc = wmma16x16x32(a, b, oacc);
      }
      int ncol = wave * 16 + (lane & 15);
#pragma unroll
      for (int r = 0; r < 8; ++r) {
        int bi = mtile * 16 + m0 + r;
        out[((size_t)bi * CFC_T + t) * CFC_OUT + ncol] = oacc[r] + biasP2;
      }
    }
  }
}

// ---------------- launcher ---------------------------------------------------
extern "C" void kernel_launch(void* const* d_in, const int* in_sizes, int n_in,
                              void* d_out, int out_size, void* d_ws, size_t ws_size,
                              hipStream_t stream) {
  const float* x   = (const float*)d_in[0];
  const float* Wb  = (const float*)d_in[1];
  const float* bb  = (const float*)d_in[2];
  const float* W1  = (const float*)d_in[3];
  const float* b1  = (const float*)d_in[4];
  const float* W2  = (const float*)d_in[5];
  const float* b2  = (const float*)d_in[6];
  const float* Wa  = (const float*)d_in[7];
  const float* ba  = (const float*)d_in[8];
  const float* Wtb = (const float*)d_in[9];
  const float* btb = (const float*)d_in[10];
  const float* Wp1 = (const float*)d_in[11];
  const float* bp1 = (const float*)d_in[12];
  const float* Wp2 = (const float*)d_in[13];
  const float* bp2 = (const float*)d_in[14];
  float* out = (float*)d_out;

  _Float16* wsH      = (_Float16*)d_ws;
  _Float16* wbPack   = wsH;
  _Float16* headPack = wbPack + WB_HALVES;
  _Float16* wp1Pack  = headPack + HEAD_HALVES;
  _Float16* wp2Pack  = wp1Pack + WP1_HALVES;

  // pack weights into WMMA B-fragment layout (f16), fold Wa+Wtb
  pack_b_f16<<<dim3((WB_HALVES + 255) / 256), dim3(256), 0, stream>>>(
      Wb, wbPack, 320, 128);
  pack_heads_f16<<<dim3((HEAD_HALVES + 255) / 256), dim3(256), 0, stream>>>(
      W1, W2, Wa, Wtb, headPack);
  pack_b_f16<<<dim3((WP1_HALVES + 255) / 256), dim3(256), 0, stream>>>(
      Wp1, wp1Pack, 256, 128);
  pack_b_f16<<<dim3((WP2_HALVES + 255) / 256), dim3(256), 0, stream>>>(
      Wp2, wp2Pack, 128, 64);

  // fused sequential scan + projection: 16 batch tiles, 8 waves each,
  // 208KB LDS per workgroup (head weights LDS-resident, Wb + h-A-frags in VGPRs).
  cfc_scan_kernel<<<dim3(16), dim3(256), LDS_BYTES, stream>>>(
      x, bb, b1, b2, ba, btb, bp1, bp2,
      wbPack, headPack, wp1Pack, wp2Pack, out);
}
